// AttentionRoutingDetector_89343909692190
// MI455X (gfx1250) — compile-verified, matrix operand order checked by reading, not run on hardware
//
#include <hip/hip_runtime.h>
#include <hip/hip_bf16.h>

// ---------------------------------------------------------------------------
// CDNA5 (gfx1250) wave32 + WMMA implementation of AttentionRoutingDetector.
// All GEMM-shaped work uses v_wmma_f32_16x16x32_bf16; f32 activations are
// converted to bf16 while staging into LDS.
// GEMM: BM=128, BN=64, BK=64 -> 8 WMMAs per wave per barrier pair.
// ---------------------------------------------------------------------------

typedef __attribute__((ext_vector_type(16))) __bf16 v16bf;
typedef __attribute__((ext_vector_type(8)))  float  v8f;

__device__ inline unsigned short f2bf(float x) {
  unsigned int u = __float_as_uint(x);
  unsigned int r = (u + 0x7FFFu + ((u >> 16) & 1u)) >> 16;  // RNE
  return (unsigned short)r;
}

union FragU { uint4 q[2]; v16bf v; };

// A/B fragment per ISA 16-bit A-matrix 16x32 layout: lane L holds row (L&15),
// K-base (L>>4)*8; VGPR0-3 <- K[kb..kb+7], VGPR4-7 <- K[16+kb..16+kb+7].
// Both runs are 8 contiguous halfs -> two ds_load_b128.
__device__ inline v16bf load_frag(const unsigned short* rowbase, int kb) {
  FragU f;
  f.q[0] = *(const uint4*)(rowbase + kb);
  f.q[1] = *(const uint4*)(rowbase + 16 + kb);
  return f.v;
}

__device__ inline v8f wmma_bf16(v16bf a, v16bf b, v8f c) {
  return __builtin_amdgcn_wmma_f32_16x16x32_bf16(false, a, false, b, (short)0, c,
                                                 false, false);
}

// ---------------------------------------------------------------------------
// Generic bf16 WMMA GEMM: C[M,N] = epi(A[M,K] @ W[K,N] + bias).
// Block tile 128x64, BK=64, 256 threads = 8 waves; wave w owns M-strip w*16 and
// all four 16-wide N tiles (4 accumulators, 8 WMMAs per K-slab).
// Requires: M % 128 == 0, K % 64 == 0 (true for all call sites).
// EPI: 0 plain, 1 *mask[row], 2 C += val*(1-mask[row]).
// ---------------------------------------------------------------------------
template <bool RELU, int EPI>
__global__ __launch_bounds__(256) void gemm_kernel(
    const float* __restrict__ A, const float* __restrict__ W,
    const float* __restrict__ bias, float* __restrict__ C,
    const float* __restrict__ mask, int M, int N, int K) {
  __shared__ __align__(16) unsigned short sA[128 * 72];  // [m][k], padded stride
  __shared__ __align__(16) unsigned short sB[64 * 72];   // [n][k] (W transposed)
  const int tid  = threadIdx.x;
  const int bm   = blockIdx.y * 128;
  const int bn   = blockIdx.x * 64;
  const int lane = tid & 31;
  const int wv   = tid >> 5;
  const int mt   = wv * 16;
  const int kb   = (lane >> 4) * 8;
  const int mA   = mt + (lane & 15);

  v8f acc[4] = {};

  for (int k0 = 0; k0 < K; k0 += 64) {
    // Stage A slab (128x64), f32 -> bf16; coalesced on k.
#pragma unroll
    for (int i = 0; i < 32; ++i) {
      int lin = tid + i * 256;
      int row = lin >> 6, col = lin & 63;
      sA[row * 72 + col] = f2bf(A[(size_t)(bm + row) * K + (k0 + col)]);
    }
    // Stage W slab transposed (sB[n][k]); coalesced on k within a row.
#pragma unroll
    for (int i = 0; i < 16; ++i) {
      int lin = tid + i * 256;
      int n = lin >> 6, kc = lin & 63;
      int gn = bn + n;
      float w = (gn < N) ? W[(size_t)(k0 + kc) * N + gn] : 0.0f;
      sB[n * 72 + kc] = f2bf(w);
    }
    // Prefetch next K-slab into caches while this slab computes.
    if (k0 + 64 < K) {
      __builtin_prefetch(&A[(size_t)(bm + (tid & 127)) * K + (k0 + 64)], 0, 3);
      __builtin_prefetch(&W[(size_t)(k0 + 64 + (tid & 63)) * N + bn], 0, 3);
    }
    __syncthreads();
#pragma unroll
    for (int kk = 0; kk < 2; ++kk) {
      v16bf a = load_frag(sA + mA * 72 + kk * 32, kb);
#pragma unroll
      for (int t = 0; t < 4; ++t) {
        v16bf b = load_frag(sB + (t * 16 + (lane & 15)) * 72 + kk * 32, kb);
        acc[t] = wmma_bf16(a, b, acc[t]);
      }
    }
    __syncthreads();
  }

  // Epilogue. C/D layout: lane holds n=(lane&15); VGPR v -> m = v + 8*(lane>>4).
#pragma unroll
  for (int v = 0; v < 8; ++v) {
    int row = bm + mt + v + 8 * (lane >> 4);
    float mk = 0.0f;
    if (EPI != 0) mk = mask[row];
#pragma unroll
    for (int t = 0; t < 4; ++t) {
      int gn = bn + t * 16 + (lane & 15);
      if (gn < N) {
        float val = acc[t][v] + bias[gn];
        if (RELU) val = fmaxf(val, 0.0f);
        size_t idx = (size_t)row * N + gn;
        if (EPI == 0)      C[idx] = val;
        else if (EPI == 1) C[idx] = val * mk;
        else               C[idx] = C[idx] + val * (1.0f - mk);
      }
    }
  }
}

// ---------------------------------------------------------------------------
// Fused backbone: conv1(3->64,s2)+relu -> conv2(64->64,s2 via WMMA im2col)
// +relu -> 8x8 avg pool. One WG per (b, gy, gx) pooled cell; writes f(B,64,16,16).
// ---------------------------------------------------------------------------
__global__ __launch_bounds__(256) void conv_fused_kernel(
    const float* __restrict__ images, const float* __restrict__ cw1,
    const float* __restrict__ cb1, const float* __restrict__ cw2,
    const float* __restrict__ cb2, float* __restrict__ f_out) {
  __shared__ __align__(16) unsigned short sIm[64 * 584];  // im2col 64px x 576 bf16
  __shared__ __align__(16) unsigned short sWT[64 * 40];   // cw2 K-step tile [co][k]
  __shared__ float sW[3 * 35 * 36 + 64 * 27];             // image tile + cw1
  __shared__ float sRed[64];

  const int tid = threadIdx.x;
  const int gx = blockIdx.x, gy = blockIdx.y, b = blockIdx.z;

  // Phase 1a: stage 35x35x3 image region (stride-2 receptive field of the cell).
  for (int lin = tid; lin < 3 * 35 * 36; lin += 256) {
    int c = lin / (35 * 36), rem = lin % (35 * 36);
    int y = rem / 36, x = rem % 36;
    float v = 0.0f;
    int Y = gy * 32 + y, X = gx * 32 + x;
    if (x < 35 && Y < 512 && X < 512)
      v = images[((size_t)(b * 3 + c) * 512 + Y) * 512 + X];
    sW[lin] = v;
  }
  for (int lin = tid; lin < 64 * 27; lin += 256) sW[3 * 35 * 36 + lin] = cw1[lin];
  if (tid < 64) sRed[tid] = 0.0f;
  __syncthreads();

  // Phase 1b: conv1 over 17x17x64 tile, relu, scatter into bf16 im2col in LDS.
  for (int lin = tid; lin < 64 * 289; lin += 256) {
    int co = lin & 63, p = lin >> 6;
    int y1 = p / 17, x1 = p % 17;
    float acc = cb1[co];
    const float* wb = &sW[3 * 35 * 36 + co * 27];
#pragma unroll
    for (int ci = 0; ci < 3; ++ci)
#pragma unroll
      for (int ky = 0; ky < 3; ++ky)
#pragma unroll
        for (int kx = 0; kx < 3; ++kx)
          acc += sW[(ci * 35 + 2 * y1 + ky) * 36 + (2 * x1 + kx)] *
                 wb[ci * 9 + ky * 3 + kx];
    bool inR = (gy * 16 + y1 < 256) && (gx * 16 + x1 < 256);
    unsigned short bv = f2bf(inR ? fmaxf(acc, 0.0f) : 0.0f);
    // im2col scatter: value lands at pixel (y2,x2) slot (ky,kx) with y1=2*y2+ky.
    for (int ky = 0; ky < 3; ++ky) {
      int yy = y1 - ky;
      if (yy < 0 || (yy & 1) || yy > 14) continue;
      int y2 = yy >> 1;
      for (int kx = 0; kx < 3; ++kx) {
        int xx = x1 - kx;
        if (xx < 0 || (xx & 1) || xx > 14) continue;
        int x2 = xx >> 1;
        sIm[(y2 * 8 + x2) * 584 + co * 9 + ky * 3 + kx] = bv;
      }
    }
  }

  // Phase 2: conv2 as WMMA GEMM: D[64px, 64co] = im2col[64,576] @ cw2^T[576,64].
  const int lane = tid & 31, wv = tid >> 5;
  const int mt = (wv & 3) * 16, ntb = (wv >> 2) * 32;
  const int kb = (lane >> 4) * 8;
  const int mA = mt + (lane & 15);
  v8f c0 = {}, c1 = {};
  for (int k0 = 0; k0 < 576; k0 += 32) {
    // cw2 is [co][ci*9+kk] = [co][576] contiguous: stage [co][kc] directly.
#pragma unroll
    for (int i = 0; i < 8; ++i) {
      int lin = tid + i * 256;
      int kc = lin & 31, n = lin >> 5;
      sWT[n * 40 + kc] = f2bf(cw2[(size_t)n * 576 + k0 + kc]);
    }
    __syncthreads();
    v16bf a  = load_frag(sIm + mA * 584 + k0, kb);
    v16bf b0 = load_frag(sWT + (ntb + (lane & 15)) * 40, kb);
    v16bf b1 = load_frag(sWT + (ntb + 16 + (lane & 15)) * 40, kb);
    c0 = wmma_bf16(a, b0, c0);
    c1 = wmma_bf16(a, b1, c1);
    __syncthreads();
  }

  // relu + 8x8 average pool (sum over the 64 pixels of each out-channel).
  float bias0 = cb2[ntb + (lane & 15)];
  float bias1 = cb2[ntb + 16 + (lane & 15)];
  float p0 = 0.0f, p1 = 0.0f;
#pragma unroll
  for (int v = 0; v < 8; ++v) {
    p0 += fmaxf(c0[v] + bias0, 0.0f);
    p1 += fmaxf(c1[v] + bias1, 0.0f);
  }
  atomicAdd(&sRed[ntb + (lane & 15)], p0);
  atomicAdd(&sRed[ntb + 16 + (lane & 15)], p1);
  __syncthreads();
  if (tid < 64)
    f_out[((b * 64 + tid) * 16 + gy) * 16 + gx] = sRed[tid] * (1.0f / 64.0f);
}

// ---------------------------------------------------------------------------
// Attention head + router: token MLP 64->32->1, sigmoid, hard mask (score>thr).
// ---------------------------------------------------------------------------
__global__ __launch_bounds__(256) void attention_kernel(
    const float* __restrict__ f, const float* __restrict__ aw1,
    const float* __restrict__ ab1, const float* __restrict__ aw2,
    const float* __restrict__ ab2, const float* __restrict__ thr,
    float* __restrict__ scores, float* __restrict__ maskp) {
  __shared__ float s_aw1[64 * 32];
  __shared__ float s_ab1[32];
  __shared__ float s_aw2[32];
  const int tid = threadIdx.x;
  for (int i = tid; i < 2048; i += 256) s_aw1[i] = aw1[i];
  if (tid < 32) { s_ab1[tid] = ab1[tid]; s_aw2[tid] = aw2[tid]; }
  __syncthreads();
  int r = blockIdx.x * 256 + tid;  // 8192 tokens
  int b = r >> 8, p = r & 255;
  float h[32];
#pragma unroll
  for (int j = 0; j < 32; ++j) h[j] = s_ab1[j];
  for (int c = 0; c < 64; ++c) {
    float fv = f[b * 16384 + c * 256 + p];
#pragma unroll
    for (int j = 0; j < 32; ++j) h[j] += fv * s_aw1[c * 32 + j];
  }
  float s = ab2[0];
#pragma unroll
  for (int j = 0; j < 32; ++j) s += fmaxf(h[j], 0.0f) * s_aw2[j];
  float sc = 1.0f / (1.0f + __expf(-s));
  scores[r] = sc;
  maskp[r]  = (sc > thr[0]) ? 1.0f : 0.0f;  // forward value of straight-through
}

// ---------------------------------------------------------------------------
// Aggregator: mean pool + attention pool over 256 tokens -> 256 -> g = relu(gw).
// One WG per batch element.
// ---------------------------------------------------------------------------
__global__ __launch_bounds__(256) void aggregator_kernel(
    const float* __restrict__ comb, const float* __restrict__ scores,
    const float* __restrict__ gw, const float* __restrict__ gb,
    float* __restrict__ g) {
  __shared__ float sS[256];
  __shared__ float sPool[256];
  __shared__ float sSum;
  const int tid = threadIdx.x, b = blockIdx.x;
  sS[tid] = scores[b * 256 + tid];
  __syncthreads();
  if (tid == 0) {
    float t = 0.0f;
    for (int i = 0; i < 256; ++i) t += sS[i];
    sSum = t + 1e-6f;
  }
  __syncthreads();
  if (tid < 128) {
    float acc = 0.0f;
    for (int p = 0; p < 256; ++p) acc += comb[(size_t)(b * 256 + p) * 128 + tid];
    sPool[tid] = acc * (1.0f / 256.0f);
  } else {
    int d = tid - 128;
    float acc = 0.0f;
    for (int p = 0; p < 256; ++p)
      acc += (sS[p] / sSum) * comb[(size_t)(b * 256 + p) * 128 + d];
    sPool[tid] = acc;
  }
  __syncthreads();
  float acc = gb[tid];
  for (int k = 0; k < 256; ++k) acc += sPool[k] * gw[k * 256 + tid];
  g[b * 256 + tid] = fmaxf(acc, 0.0f);
}

// fuse[r][k] = k<256 ? g[b][k] : comb[r][k-256]  (8192 x 384)
__global__ __launch_bounds__(256) void fuse_kernel(const float* __restrict__ g,
                                                   const float* __restrict__ comb,
                                                   float* __restrict__ fuse) {
  int idx = blockIdx.x * 256 + threadIdx.x;
  if (idx >= 8192 * 384) return;
  int r = idx / 384, k = idx % 384;
  int b = r >> 8;
  fuse[idx] = (k < 256) ? g[b * 256 + k] : comb[(size_t)r * 128 + (k - 256)];
}

// reg head: N=4 -> plain VALU kernel, one token per thread.
__global__ __launch_bounds__(256) void reg_kernel(const float* __restrict__ hd,
                                                  const float* __restrict__ reg_w,
                                                  const float* __restrict__ reg_b,
                                                  float* __restrict__ out) {
  int r = blockIdx.x * 256 + threadIdx.x;  // 8192
  float a0 = reg_b[0], a1 = reg_b[1], a2 = reg_b[2], a3 = reg_b[3];
  for (int k = 0; k < 256; ++k) {
    float hv = hd[(size_t)r * 256 + k];
    a0 += hv * reg_w[k * 4 + 0];
    a1 += hv * reg_w[k * 4 + 1];
    a2 += hv * reg_w[k * 4 + 2];
    a3 += hv * reg_w[k * 4 + 3];
  }
  out[r * 4 + 0] = a0; out[r * 4 + 1] = a1;
  out[r * 4 + 2] = a2; out[r * 4 + 3] = a3;
}

// ---------------------------------------------------------------------------
extern "C" void kernel_launch(void* const* d_in, const int* in_sizes, int n_in,
                              void* d_out, int out_size, void* d_ws, size_t ws_size,
                              hipStream_t stream) {
  const float* images = (const float*)d_in[0];
  const float* patches = (const float*)d_in[1];  // (32,256,3072) contiguous
  const float* cw1 = (const float*)d_in[2];
  const float* cb1 = (const float*)d_in[3];
  const float* cw2 = (const float*)d_in[4];
  const float* cb2 = (const float*)d_in[5];
  const float* aw1 = (const float*)d_in[6];
  const float* ab1 = (const float*)d_in[7];
  const float* aw2 = (const float*)d_in[8];
  const float* ab2 = (const float*)d_in[9];
  const float* thr = (const float*)d_in[10];
  const float* bw1 = (const float*)d_in[11]; const float* bb1 = (const float*)d_in[12];
  const float* bw2 = (const float*)d_in[13]; const float* bb2 = (const float*)d_in[14];
  const float* bw3 = (const float*)d_in[15]; const float* bb3 = (const float*)d_in[16];
  const float* bw4 = (const float*)d_in[17]; const float* bb4 = (const float*)d_in[18];
  const float* sw1 = (const float*)d_in[19]; const float* sb1 = (const float*)d_in[20];
  const float* sw2 = (const float*)d_in[21]; const float* sb2 = (const float*)d_in[22];
  const float* gw  = (const float*)d_in[23]; const float* gb  = (const float*)d_in[24];
  const float* dw1 = (const float*)d_in[25]; const float* db1 = (const float*)d_in[26];
  const float* cls_w = (const float*)d_in[27]; const float* cls_b = (const float*)d_in[28];
  const float* reg_w = (const float*)d_in[29]; const float* reg_b = (const float*)d_in[30];
  (void)in_sizes; (void)n_in; (void)out_size; (void)ws_size;

  // Workspace carve-out with buffer reuse (~41 MB total).
  char* ws = (char*)d_ws;
  size_t off = 0;
  auto carve = [&](size_t bytes) -> float* {
    float* p = (float*)(ws + off);
    off = (off + bytes + 255) & ~(size_t)255;
    return p;
  };
  float* f      = carve((size_t)32 * 64 * 16 * 16 * 4);  // pooled features
  float* scores = carve(8192 * 4);
  float* maskp  = carve(8192 * 4);
  float* g      = carve(32 * 256 * 4);
  float* comb   = carve((size_t)8192 * 128 * 4);
  float* hs     = carve((size_t)8192 * 64 * 4);
  float* bufA   = carve((size_t)8192 * 512 * 4);
  float* bufB   = carve((size_t)8192 * 512 * 4);
  float* h1 = bufA;        // big MLP layer 1
  float* h2 = bufB;        // layer 2
  float* h3 = bufA;        // layer 3 (h1 dead)
  float* fuse = bufB;      // 8192x384 (h2 dead)
  float* hd = bufA;        // 8192x256 (h3 dead)

  dim3 blk(256);
  conv_fused_kernel<<<dim3(16, 16, 32), blk, 0, stream>>>(images, cw1, cb1, cw2,
                                                          cb2, f);
  attention_kernel<<<dim3(32), blk, 0, stream>>>(f, aw1, ab1, aw2, ab2, thr,
                                                 scores, maskp);
  // Big expert MLP: 3072->512->512->512->128, masked output into comb.
  gemm_kernel<true, 0><<<dim3(8, 64), blk, 0, stream>>>(patches, bw1, bb1, h1,
                                                        nullptr, 8192, 512, 3072);
  gemm_kernel<true, 0><<<dim3(8, 64), blk, 0, stream>>>(h1, bw2, bb2, h2,
                                                        nullptr, 8192, 512, 512);
  gemm_kernel<true, 0><<<dim3(8, 64), blk, 0, stream>>>(h2, bw3, bb3, h3,
                                                        nullptr, 8192, 512, 512);
  gemm_kernel<false, 1><<<dim3(2, 64), blk, 0, stream>>>(h3, bw4, bb4, comb,
                                                         maskp, 8192, 128, 512);
  // Small expert MLP: 3072->64->128, (1-mask)-weighted accumulate into comb.
  gemm_kernel<true, 0><<<dim3(1, 64), blk, 0, stream>>>(patches, sw1, sb1, hs,
                                                        nullptr, 8192, 64, 3072);
  gemm_kernel<false, 2><<<dim3(2, 64), blk, 0, stream>>>(hs, sw2, sb2, comb,
                                                         maskp, 8192, 128, 64);
  // Aggregate + broadcast + detection trunk.
  aggregator_kernel<<<dim3(32), blk, 0, stream>>>(comb, scores, gw, gb, g);
  fuse_kernel<<<dim3(12288), blk, 0, stream>>>(g, comb, fuse);
  gemm_kernel<true, 0><<<dim3(4, 64), blk, 0, stream>>>(fuse, dw1, db1, hd,
                                                        nullptr, 8192, 256, 384);
  // Heads: cls (8192x80) -> d_out[0..], reg (8192x4) -> d_out[655360..].
  float* cls_out = (float*)d_out;
  float* reg_out = (float*)d_out + (size_t)8192 * 80;
  gemm_kernel<false, 0><<<dim3(2, 64), blk, 0, stream>>>(hd, cls_w, cls_b,
                                                         cls_out, nullptr, 8192,
                                                         80, 256);
  reg_kernel<<<dim3(32), blk, 0, stream>>>(hd, reg_w, reg_b, reg_out);
}
